// DeepMamba_29824252903748
// MI455X (gfx1250) — matrix operand order repaired and checked by reading, not compiled, not run on hardware
//
#include <hip/hip_runtime.h>
#include <hip/hip_bf16.h>

// ---------------- problem constants (match reference) ----------------
#define LAYERS  4
#define DMODEL  1024
#define DSTATE  16
#define DCONV   4
#define DINNER  2048
#define DTRANK  64
#define BB      4
#define TT      2048
#define MROWS   (BB * TT)          // 8192 rows for all GEMMs

typedef __attribute__((ext_vector_type(16))) __bf16          v16bf;
typedef __attribute__((ext_vector_type(8)))  float           v8f;

// round-to-nearest-even f32 -> bf16 (bit pattern)
static __device__ __forceinline__ unsigned short f2bf(float f) {
  unsigned u = __builtin_bit_cast(unsigned, f);
  u += 0x7FFFu + ((u >> 16) & 1u);
  return (unsigned short)(u >> 16);
}

union Frag16 {                      // 16 bf16 = two 16-byte quads = 8 VGPRs
  uint4 q[2];
  v16bf v;
};

// fragment gather: pure b128 loads per the ISA 16-bit striping
//   A: lanes 0-15 hold K 0..7 / 16..23, lanes 16-31 hold K 8..15 / 24..31
//   B: lanes 0-15 -> N=lane, K 0..7 / 16..23 ; lanes 16-31 -> N=lane-16, K 8..15 / 24..31
template <int NT>
static __device__ __forceinline__ void load_frags(
    const unsigned short* __restrict__ Arow,
    const unsigned short* const (&Wrow)[NT],
    int kk, Frag16& fa, Frag16 (&fb)[NT])
{
  fa.q[0] = *(const uint4*)(Arow + kk);        // K hh*8 .. hh*8+7
  fa.q[1] = *(const uint4*)(Arow + kk + 16);   // K 16+hh*8 .. +7
#pragma unroll
  for (int nt = 0; nt < NT; ++nt) {
    fb[nt].q[0] = *(const uint4*)(Wrow[nt] + kk);      // K hh*8 .. hh*8+7
    fb[nt].q[1] = *(const uint4*)(Wrow[nt] + kk + 16); // K 16+hh*8 .. +7
  }
}

template <int NT>
static __device__ __forceinline__ void mma_all(
    const Frag16& fa, const Frag16 (&fb)[NT], v8f (&acc)[NT])
{
#pragma unroll
  for (int nt = 0; nt < NT; ++nt)
    acc[nt] = __builtin_amdgcn_wmma_f32_16x16x32_bf16(
        false, fa.v, false, fb[nt].v, (short)0, acc[nt], false, false);
}

// =====================================================================
// GEMM: C[M, NT*16*gridX] = A[M,K] * W[N,K]^T ; A,W bf16, C f32.
// NT = compile-time N-tile count (branch-free body).
// Software-pipelined k-loop (step 64, two fragment sets) so loads for
// the next half-step are in flight while the current WMMAs execute.
// Requires K % 64 == 0 (holds: 1024 / 2048 / 64).
// =====================================================================
template <int NT>
__global__ __launch_bounds__(256) void gemm_bf16_wmma(
    const unsigned short* __restrict__ A, const unsigned short* __restrict__ W,
    float* __restrict__ C, int M, int K,
    int lda, int ldw, int ldc, int colbase)
{
  const int lane = threadIdx.x & 31;
  const int wave = threadIdx.x >> 5;
  const int row0 = blockIdx.y * 128 + wave * 16;
  const int col0 = colbase + blockIdx.x * (NT * 16);
  if (row0 >= M) return;                       // wave-uniform

  const int hh = lane >> 4;                    // 0: lanes 0-15, 1: lanes 16-31
  const int ml = lane & 15;

  v8f acc[NT];
#pragma unroll
  for (int nt = 0; nt < NT; ++nt) acc[nt] = v8f{};

  const unsigned short* Arow = A + (size_t)(row0 + ml) * lda + hh * 8;
  const unsigned short* Wrow[NT];
#pragma unroll
  for (int nt = 0; nt < NT; ++nt)
    Wrow[nt] = W + (size_t)(col0 + nt * 16 + ml) * ldw + hh * 8;

  Frag16 faA, faB;
  Frag16 fbA[NT], fbB[NT];
  load_frags<NT>(Arow, Wrow, 0, faA, fbA);

  for (int kk = 0; kk < K; kk += 64) {
    load_frags<NT>(Arow, Wrow, kk + 32, faB, fbB);   // prefetch half-step B set
    if (kk + 128 < K)                                // uniform guard
      __builtin_prefetch(Arow + kk + 128, 0, 1);
    mma_all<NT>(faA, fbA, acc);                      // waits only on older loads
    if (kk + 64 < K)                                 // uniform guard
      load_frags<NT>(Arow, Wrow, kk + 64, faA, fbA);
    mma_all<NT>(faB, fbB, acc);
  }

  // store: C/D layout: VGPR r -> (lanes 0-15: M=r, lanes 16-31: M=8+r), N=lane&15
#pragma unroll
  for (int nt = 0; nt < NT; ++nt) {
    int n = col0 + nt * 16 + ml;
#pragma unroll
    for (int r = 0; r < 8; ++r) {
      int m = row0 + hh * 8 + r;
      C[(size_t)m * ldc + n] = acc[nt][r];
    }
  }
}

// =====================================================================
// f32 -> bf16 bulk convert (4 elements / thread), n multiple of 4
// =====================================================================
__global__ __launch_bounds__(256) void cvt_bf16_k(
    const float* __restrict__ in, unsigned short* __restrict__ out, int n4)
{
  int i = blockIdx.x * 256 + threadIdx.x;
  if (i >= n4) return;
  float4 v = *(const float4*)(in + (size_t)i * 4);
  ushort4 o;
  o.x = f2bf(v.x); o.y = f2bf(v.y); o.z = f2bf(v.z); o.w = f2bf(v.w);
  *(ushort4*)(out + (size_t)i * 4) = o;
}

// =====================================================================
// LayerNorm: one wave per 1024-element row (8 waves / block); bf16 out
// =====================================================================
__global__ __launch_bounds__(256) void layernorm_k(
    const float* __restrict__ x, const float* __restrict__ w,
    const float* __restrict__ b, unsigned short* __restrict__ out, int rows)
{
  int wave = threadIdx.x >> 5, lane = threadIdx.x & 31;
  int row = blockIdx.x * 8 + wave;
  if (row >= rows) return;
  const float* xr = x + (size_t)row * DMODEL;

  float4 v[8];
  float s = 0.f, s2 = 0.f;
#pragma unroll
  for (int i = 0; i < 8; ++i) {
    v[i] = *(const float4*)(xr + i * 128 + lane * 4);
    s  += v[i].x + v[i].y + v[i].z + v[i].w;
    s2 += v[i].x * v[i].x + v[i].y * v[i].y + v[i].z * v[i].z + v[i].w * v[i].w;
  }
#pragma unroll
  for (int o = 16; o; o >>= 1) {
    s  += __shfl_xor(s,  o, 32);
    s2 += __shfl_xor(s2, o, 32);
  }
  float mu  = s  * (1.f / DMODEL);
  float var = s2 * (1.f / DMODEL) - mu * mu;
  float rs  = rsqrtf(var + 1e-5f);

  unsigned short* orow = out + (size_t)row * DMODEL;
#pragma unroll
  for (int i = 0; i < 8; ++i) {
    int d0 = i * 128 + lane * 4;
    ushort4 o;
    o.x = f2bf((v[i].x - mu) * rs * w[d0 + 0] + b[d0 + 0]);
    o.y = f2bf((v[i].y - mu) * rs * w[d0 + 1] + b[d0 + 1]);
    o.z = f2bf((v[i].z - mu) * rs * w[d0 + 2] + b[d0 + 2]);
    o.w = f2bf((v[i].w - mu) * rs * w[d0 + 3] + b[d0 + 3]);
    *(ushort4*)(orow + d0) = o;
  }
}

// =====================================================================
// Depthwise causal conv (k=4) on x-half of xz, + bias + SiLU
// writes f32 (scan input) and bf16 (GEMM A operand)
// =====================================================================
__global__ __launch_bounds__(256) void conv_silu_k(
    const float* __restrict__ xz, const float* __restrict__ cw,
    const float* __restrict__ cb, float* __restrict__ xc,
    unsigned short* __restrict__ xcbf)
{
  int idx = blockIdx.x * 256 + threadIdx.x;
  if (idx >= MROWS * DINNER) return;
  int d  = idx % DINNER;
  int bt = idx / DINNER;
  int t  = bt % TT;
  int bb = bt / TT;
  float acc = cb[d];
#pragma unroll
  for (int j = 0; j < DCONV; ++j) {
    int ts = t - (DCONV - 1) + j;
    if (ts >= 0)
      acc += xz[((size_t)(bb * TT + ts)) * (2 * DINNER) + d] * cw[d * DCONV + j];
  }
  float val = acc * __frcp_rn(1.f + __expf(-acc));
  xc[idx]   = val;
  xcbf[idx] = f2bf(val);
}

// softplus(x + bias) in place on dtp (f32, scan input)
__global__ __launch_bounds__(256) void softplus_bias_k(
    float* __restrict__ dtp, const float* __restrict__ dtb)
{
  int idx = blockIdx.x * 256 + threadIdx.x;
  if (idx >= MROWS * DINNER) return;
  int d = idx % DINNER;
  float x = dtp[idx] + dtb[d];
  dtp[idx] = (x > 20.f) ? x : log1pf(__expf(x));
}

// dt-slice (cols 0..63 of 96-wide x_dbl) -> bf16, lda 96 -> 64
__global__ __launch_bounds__(256) void dtslice_bf16_k(
    const float* __restrict__ xdbl, unsigned short* __restrict__ dtbf)
{
  int idx = blockIdx.x * 256 + threadIdx.x;
  if (idx >= MROWS * DTRANK) return;
  int m = idx >> 6, r = idx & 63;
  dtbf[idx] = f2bf(xdbl[(size_t)m * 96 + r]);
}

// ybf = bf16( y * silu(z) )   (z = second half of xz)
__global__ __launch_bounds__(256) void gate_k(
    const float* __restrict__ y, const float* __restrict__ xz,
    unsigned short* __restrict__ ybf)
{
  int idx = blockIdx.x * 256 + threadIdx.x;
  if (idx >= MROWS * DINNER) return;
  int d  = idx % DINNER;
  int bt = idx / DINNER;
  float z = xz[(size_t)bt * (2 * DINNER) + DINNER + d];
  ybf[idx] = f2bf(y[idx] * z * __frcp_rn(1.f + __expf(-z)));
}

// =====================================================================
// Selective scan: one lane per (b, d) channel, h[16] in VGPRs,
// B/C (shared across d) staged through LDS in 64-step chunks.
// =====================================================================
#define SCHUNK 64
__global__ __launch_bounds__(256) void scan_k(
    const float* __restrict__ u,     const float* __restrict__ delta,
    const float* __restrict__ xdbl,  const float* __restrict__ A_log,
    const float* __restrict__ Dskip, float* __restrict__ y)
{
  __shared__ float sBC[SCHUNK * 32];          // [tt][0..15]=Bm, [tt][16..31]=C
  int bb = blockIdx.x >> 3;
  int d  = (blockIdx.x & 7) * 256 + threadIdx.x;

  float a[DSTATE], h[DSTATE];
#pragma unroll
  for (int n = 0; n < DSTATE; ++n) {
    a[n] = -__expf(A_log[d * DSTATE + n]);
    h[n] = 0.f;
  }
  float Dv = Dskip[d];

  for (int t0 = 0; t0 < TT; t0 += SCHUNK) {
    __syncthreads();
    for (int i = threadIdx.x; i < SCHUNK * 32; i += 256) {
      int tt = i >> 5, q = i & 31;
      sBC[i] = xdbl[((size_t)(bb * TT + t0 + tt)) * 96 + 64 + q];
    }
    __syncthreads();
    for (int tt = 0; tt < SCHUNK; ++tt) {
      size_t off = ((size_t)(bb * TT + t0 + tt)) * DINNER + d;
      float dl = delta[off];
      float uv = u[off];
      float du = dl * uv;
      float yt = 0.f;
      const float* bc = &sBC[tt * 32];
#pragma unroll
      for (int n = 0; n < DSTATE; ++n) {
        h[n] = __expf(dl * a[n]) * h[n] + du * bc[n];
        yt  += h[n] * bc[16 + n];
      }
      y[off] = yt + uv * Dv;
    }
  }
}

// (b,t,d) -> (t,b,d)
__global__ __launch_bounds__(256) void transpose_out_k(
    const float* __restrict__ xin, float* __restrict__ out)
{
  int idx = blockIdx.x * 256 + threadIdx.x;
  if (idx >= MROWS * DMODEL) return;
  int dd = idx % DMODEL;
  int bt = idx / DMODEL;
  int t  = bt % TT;
  int bb = bt / TT;
  out[((size_t)t * BB + bb) * DMODEL + dd] = xin[idx];
}

// =====================================================================
extern "C" void kernel_launch(void* const* d_in, const int* in_sizes, int n_in,
                              void* d_out, int out_size, void* d_ws, size_t ws_size,
                              hipStream_t stream) {
  (void)in_sizes; (void)n_in; (void)out_size; (void)ws_size;

  const float* x_in   = (const float*)d_in[0];
  const float* ln_w   = (const float*)d_in[1];
  const float* ln_b   = (const float*)d_in[2];
  const float* inpw   = (const float*)d_in[3];
  const float* conv_w = (const float*)d_in[4];
  const float* conv_b = (const float*)d_in[5];
  const float* xpw    = (const float*)d_in[6];
  const float* dtw    = (const float*)d_in[7];
  const float* dtb    = (const float*)d_in[8];
  const float* A_log  = (const float*)d_in[9];
  const float* Dsk    = (const float*)d_in[10];
  const float* outw   = (const float*)d_in[11];

  // ---- workspace layout (f32 region, then bf16 region) ----
  float* ws = (float*)d_ws;
  size_t o = 0;
  float* x_cur = ws + o; o += (size_t)MROWS * DMODEL;       // next-layer input
  float* xz    = ws + o; o += (size_t)MROWS * 2 * DINNER;
  float* xc    = ws + o; o += (size_t)MROWS * DINNER;
  float* xdbl  = ws + o; o += (size_t)MROWS * 96;
  float* dtp   = ws + o; o += (size_t)MROWS * DINNER;       // -> delta in place
  float* ybuf  = ws + o; o += (size_t)MROWS * DINNER;

  unsigned short* hs = (unsigned short*)(ws + o);
  size_t h = 0;
  unsigned short* xln_bf = hs + h; h += (size_t)MROWS * DMODEL;
  unsigned short* xc_bf  = hs + h; h += (size_t)MROWS * DINNER;
  unsigned short* dt_bf  = hs + h; h += (size_t)MROWS * DTRANK;
  unsigned short* y_bf   = hs + h; h += (size_t)MROWS * DINNER;
  unsigned short* win_bf = hs + h; h += (size_t)LAYERS * 2 * DINNER * DMODEL;
  unsigned short* wxp_bf = hs + h; h += (size_t)LAYERS * 96 * DINNER;
  unsigned short* wdt_bf = hs + h; h += (size_t)LAYERS * DINNER * DTRANK;
  unsigned short* wou_bf = hs + h; h += (size_t)LAYERS * DMODEL * DINNER;

  // ---- one-time (per launch) weight conversions to bf16 ----
  {
    int n4;
    n4 = (LAYERS * 2 * DINNER * DMODEL) / 4;
    cvt_bf16_k<<<(n4 + 255) / 256, 256, 0, stream>>>(inpw, win_bf, n4);
    n4 = (LAYERS * 96 * DINNER) / 4;
    cvt_bf16_k<<<(n4 + 255) / 256, 256, 0, stream>>>(xpw, wxp_bf, n4);
    n4 = (LAYERS * DINNER * DTRANK) / 4;
    cvt_bf16_k<<<(n4 + 255) / 256, 256, 0, stream>>>(dtw, wdt_bf, n4);
    n4 = (LAYERS * DMODEL * DINNER) / 4;
    cvt_bf16_k<<<(n4 + 255) / 256, 256, 0, stream>>>(outw, wou_bf, n4);
  }

  const int ew_grid = (MROWS * DINNER + 255) / 256;

  for (int l = 0; l < LAYERS; ++l) {
    const float* xin = (l == 0) ? x_in : x_cur;

    layernorm_k<<<MROWS / 8, 256, 0, stream>>>(
        xin, ln_w + (size_t)l * DMODEL, ln_b + (size_t)l * DMODEL, xln_bf, MROWS);

    // in_proj: (8192,1024) x (4096,1024)^T -> xz (8192,4096)
    gemm_bf16_wmma<4><<<dim3((2 * DINNER) / 64, MROWS / 128), 256, 0, stream>>>(
        xln_bf, win_bf + (size_t)l * 2 * DINNER * DMODEL, xz,
        MROWS, DMODEL, DMODEL, DMODEL, 2 * DINNER, 0);

    conv_silu_k<<<ew_grid, 256, 0, stream>>>(
        xz, conv_w + (size_t)l * DINNER * DCONV, conv_b + (size_t)l * DINNER,
        xc, xc_bf);

    // x_proj: (8192,2048) x (96,2048)^T -> xdbl (8192,96)
    // cols 0..63 with NT=4, cols 64..95 with NT=2 (branch-free bodies)
    gemm_bf16_wmma<4><<<dim3(1, MROWS / 128), 256, 0, stream>>>(
        xc_bf, wxp_bf + (size_t)l * 96 * DINNER, xdbl,
        MROWS, DINNER, DINNER, DINNER, 96, 0);
    gemm_bf16_wmma<2><<<dim3(1, MROWS / 128), 256, 0, stream>>>(
        xc_bf, wxp_bf + (size_t)l * 96 * DINNER, xdbl,
        MROWS, DINNER, DINNER, DINNER, 96, 64);

    dtslice_bf16_k<<<(MROWS * DTRANK + 255) / 256, 256, 0, stream>>>(xdbl, dt_bf);

    // dt_proj: (8192,64) x (2048,64)^T -> dtp (8192,2048)
    gemm_bf16_wmma<4><<<dim3(DINNER / 64, MROWS / 128), 256, 0, stream>>>(
        dt_bf, wdt_bf + (size_t)l * DINNER * DTRANK, dtp,
        MROWS, DTRANK, DTRANK, DTRANK, DINNER, 0);

    softplus_bias_k<<<ew_grid, 256, 0, stream>>>(dtp, dtb + (size_t)l * DINNER);

    scan_k<<<BB * (DINNER / 256), 256, 0, stream>>>(
        xc, dtp, xdbl, A_log + (size_t)l * DINNER * DSTATE,
        Dsk + (size_t)l * DINNER, ybuf);

    gate_k<<<ew_grid, 256, 0, stream>>>(ybuf, xz, y_bf);

    // out_proj: (8192,2048) x (1024,2048)^T -> x_cur (8192,1024)
    gemm_bf16_wmma<4><<<dim3(DMODEL / 64, MROWS / 128), 256, 0, stream>>>(
        y_bf, wou_bf + (size_t)l * DMODEL * DINNER, x_cur,
        MROWS, DINNER, DINNER, DINNER, DMODEL, 0);
  }

  transpose_out_k<<<(MROWS * DMODEL + 255) / 256, 256, 0, stream>>>(
      x_cur, (float*)d_out);
}
